// Net_24515673326105
// MI455X (gfx1250) — matrix operand order repaired and checked by reading, not compiled, not run on hardware
//
#include <hip/hip_runtime.h>

// GNN message passing for MI455X (gfx1250, wave32, WMMA).
#define GNN_N 100000
#define GNN_E 1600000
#define GNN_D 64
#define GNN_H 16

typedef __attribute__((ext_vector_type(16))) _Float16 v16h;
typedef __attribute__((ext_vector_type(8)))  float    v8f;

union AFrag { v16h v; _Float16 h[16]; };
union CFrag { v8f  v; float     f[8];  };

__device__ __forceinline__ v8f wmma_f16(const AFrag& a, const AFrag& b, const CFrag& c) {
    // D = A(16x32 f16) * B(32x16 f16) + C(16x16 f32)
    return __builtin_amdgcn_wmma_f32_16x16x32_f16(
        /*neg_a=*/false, a.v, /*neg_b=*/false, b.v,
        /*c_mod=*/(short)0, c.v, /*reuse_a=*/false, /*reuse_b=*/false);
}

__device__ __forceinline__ void cvt8h(const float4& q0, const float4& q1, _Float16* dst) {
    dst[0] = (_Float16)q0.x; dst[1] = (_Float16)q0.y;
    dst[2] = (_Float16)q0.z; dst[3] = (_Float16)q0.w;
    dst[4] = (_Float16)q1.x; dst[5] = (_Float16)q1.y;
    dst[6] = (_Float16)q1.z; dst[7] = (_Float16)q1.w;
}

// ---------------------------------------------------------------------------
// Edge kernel: one wave32 per 16-edge tile, grid-stride, 2-stage software
// pipeline so the next tile's random x-row gather (L2-resident) overlaps the
// current tile's GEMM2 + scatter atomics.
//   m = relu(relu(x[src] @ Wm1 + bm1) @ Wm2 + bm2);  aggr[dst] += m
// All gather/scatter offsets are 32-bit (max 6.4M elements) so the compiler
// can use SADDR (uniform base + 32-bit voffset) addressing.
// ---------------------------------------------------------------------------
__global__ __launch_bounds__(128) void edge_msg_scatter(
    const float* __restrict__ x, const int* __restrict__ eidx,
    const float* __restrict__ Wm1, const float* __restrict__ bm1,
    const float* __restrict__ Wm2, const float* __restrict__ bm2,
    float* __restrict__ aggr)
{
    __shared__ float lds[4][16][17];   // per-wave 16x16 transpose staging
    const int lane = threadIdx.x & 31;
    const int wave = threadIdx.x >> 5;
    const int col  = lane & 15;
    const int hi   = lane >> 4;        // 0 or 1
    const int kb   = hi << 3;          // 0 or 8

    // Hoisted weight fragments (reused for every tile this wave processes).
    AFrag b1[2];                       // Wm1: 64x16, two K=32 blocks
    #pragma unroll
    for (int t = 0; t < 2; ++t)
        #pragma unroll
        for (int h = 0; h < 16; ++h)
            b1[t].h[h] = (_Float16)Wm1[(t * 32 + kb + h + (h & 8)) * GNN_H + col];

    AFrag b2[4];                       // Wm2: 16x64 -> 4 N-tiles, K padded to 32
    #pragma unroll
    for (int j = 0; j < 4; ++j)
        #pragma unroll
        for (int h = 0; h < 16; ++h)
            b2[j].h[h] = (h >= 8) ? (_Float16)0.0f
                            : (_Float16)Wm2[(kb + h) * GNN_D + j * 16 + col];

    const float bias1 = bm1[col];
    float bias2[4];
    #pragma unroll
    for (int j = 0; j < 4; ++j) bias2[j] = bm2[j * 16 + col];

    const int nTiles = GNN_E >> 4;     // 100000 exact
    const int stride = gridDim.x << 2;
    int tile = (blockIdx.x << 2) + wave;
    if (tile >= nTiles) return;        // wave-uniform

    // ---- pipeline prologue: issue first tile's gather ----
    float4 q[8];
    int4 d0, d1;
    {
        const unsigned ebase = (unsigned)tile << 4;
        const unsigned src = (unsigned)eidx[ebase + col];
        const float* xr = x + src * GNN_D + kb;          // 32-bit offset
        q[0] = ((const float4*)(xr     ))[0]; q[1] = ((const float4*)(xr +  4))[0];
        q[2] = ((const float4*)(xr + 16))[0]; q[3] = ((const float4*)(xr + 20))[0];
        q[4] = ((const float4*)(xr + 32))[0]; q[5] = ((const float4*)(xr + 36))[0];
        q[6] = ((const float4*)(xr + 48))[0]; q[7] = ((const float4*)(xr + 52))[0];
        d0 = ((const int4*)(eidx + GNN_E + ebase + kb))[0];
        d1 = ((const int4*)(eidx + GNN_E + ebase + kb))[1];
    }

    while (true) {
        // Consume prefetched data into fragments / 32-bit row offsets.
        AFrag a0, a1;
        cvt8h(q[0], q[1], a0.h); cvt8h(q[2], q[3], a0.h + 8);
        cvt8h(q[4], q[5], a1.h); cvt8h(q[6], q[7], a1.h + 8);
        unsigned rowOff[8];
        rowOff[0] = (unsigned)d0.x * GNN_D; rowOff[1] = (unsigned)d0.y * GNN_D;
        rowOff[2] = (unsigned)d0.z * GNN_D; rowOff[3] = (unsigned)d0.w * GNN_D;
        rowOff[4] = (unsigned)d1.x * GNN_D; rowOff[5] = (unsigned)d1.y * GNN_D;
        rowOff[6] = (unsigned)d1.z * GNN_D; rowOff[7] = (unsigned)d1.w * GNN_D;

        const int nextTile = tile + stride;
        const bool more = nextTile < nTiles;   // wave-uniform

        // ---- GEMM1: xj(16x64) @ Wm1(64x16), K=64 -> 2 WMMAs ----
        CFrag acc;
        #pragma unroll
        for (int i = 0; i < 8; ++i) acc.f[i] = bias1;
        acc.v = wmma_f16(a0, b1[0], acc);
        acc.v = wmma_f16(a1, b1[1], acc);

        // ---- prefetch next tile (hides gather latency under GEMM2+atomics) ----
        if (more) {
            const unsigned ebase = (unsigned)nextTile << 4;
            const unsigned src = (unsigned)eidx[ebase + col];
            const float* xr = x + src * GNN_D + kb;
            q[0] = ((const float4*)(xr     ))[0]; q[1] = ((const float4*)(xr +  4))[0];
            q[2] = ((const float4*)(xr + 16))[0]; q[3] = ((const float4*)(xr + 20))[0];
            q[4] = ((const float4*)(xr + 32))[0]; q[5] = ((const float4*)(xr + 36))[0];
            q[6] = ((const float4*)(xr + 48))[0]; q[7] = ((const float4*)(xr + 52))[0];
            d0 = ((const int4*)(eidx + GNN_E + ebase + kb))[0];
            d1 = ((const int4*)(eidx + GNN_E + ebase + kb))[1];
        }

        #pragma unroll
        for (int i = 0; i < 8; ++i) acc.f[i] = fmaxf(acc.f[i], 0.0f);

        // ---- transpose hidden (C layout -> A layout) through LDS ----
        #pragma unroll
        for (int v = 0; v < 8; ++v) lds[wave][kb + v][col] = acc.f[v];
        asm volatile("s_wait_dscnt 0" ::: "memory");  // same-wave DS ordering fence
        AFrag ah;
        #pragma unroll
        for (int h = 0; h < 8; ++h) ah.h[h] = (_Float16)lds[wave][col][kb + h];
        #pragma unroll
        for (int h = 8; h < 16; ++h) ah.h[h] = (_Float16)0.0f;  // K 16..31 pad

        // ---- GEMM2: hid(16x16) @ Wm2(16x64) -> 4 WMMAs, then f32 atomics ----
        #pragma unroll
        for (int j = 0; j < 4; ++j) {
            CFrag c;
            #pragma unroll
            for (int i = 0; i < 8; ++i) c.f[i] = bias2[j];
            c.v = wmma_f16(ah, b2[j], c);
            #pragma unroll
            for (int v = 0; v < 8; ++v)
                unsafeAtomicAdd(&aggr[rowOff[v] + (unsigned)(j * 16 + col)],
                                fmaxf(c.f[v], 0.0f));
        }

        if (!more) break;
        tile = nextTile;
    }
}

// ---------------------------------------------------------------------------
// Node kernel: one wave32 per 16-node tile.
//   x' = relu(relu([aggr|x] @ Wu1 + bu1) @ Wu2 + bu2)
// Concat is free: K blocks 0..63 read aggr, 64..127 read x. Safe in-place
// (each tile reads only its own 16 rows before writing them).
// ---------------------------------------------------------------------------
__global__ __launch_bounds__(128) void node_update(
    const float* xin, const float* __restrict__ aggr,
    const float* __restrict__ Wu1, const float* __restrict__ bu1,
    const float* __restrict__ Wu2, const float* __restrict__ bu2,
    float* xout)
{
    __shared__ float lds[4][16][17];
    const int lane = threadIdx.x & 31;
    const int wave = threadIdx.x >> 5;
    const int col  = lane & 15;
    const int hi   = lane >> 4;
    const int kb   = hi << 3;

    AFrag b1[4];                       // Wu1: 128x16, four K=32 blocks
    #pragma unroll
    for (int t = 0; t < 4; ++t)
        #pragma unroll
        for (int h = 0; h < 16; ++h)
            b1[t].h[h] = (_Float16)Wu1[(t * 32 + kb + h + (h & 8)) * GNN_H + col];

    AFrag b2[4];                       // Wu2: 16x64
    #pragma unroll
    for (int j = 0; j < 4; ++j)
        #pragma unroll
        for (int h = 0; h < 16; ++h)
            b2[j].h[h] = (h >= 8) ? (_Float16)0.0f
                            : (_Float16)Wu2[(kb + h) * GNN_D + j * 16 + col];

    const float bias1 = bu1[col];
    float bias2[4];
    #pragma unroll
    for (int j = 0; j < 4; ++j) bias2[j] = bu2[j * 16 + col];

    const int nTiles = (GNN_N + 15) >> 4;   // 6250 exact
    const int stride = gridDim.x << 2;
    for (int tile = (blockIdx.x << 2) + wave; tile < nTiles; tile += stride) {
        const unsigned nbase = (unsigned)tile << 4;
        const unsigned row = (nbase + col) * GNN_D;   // 32-bit element offset

        CFrag acc;
        #pragma unroll
        for (int i = 0; i < 8; ++i) acc.f[i] = bias1;
        #pragma unroll
        for (int t = 0; t < 4; ++t) {
            const float* base = (t < 2) ? (aggr + row + t * 32)
                                        : (xin  + row + (t - 2) * 32);
            AFrag a;
            const float4 q0 = ((const float4*)(base + kb))[0];
            const float4 q1 = ((const float4*)(base + kb))[1];
            const float4 q2 = ((const float4*)(base + kb + 16))[0];
            const float4 q3 = ((const float4*)(base + kb + 16))[1];
            cvt8h(q0, q1, a.h);
            cvt8h(q2, q3, a.h + 8);
            acc.v = wmma_f16(a, b1[t], acc);
        }
        #pragma unroll
        for (int i = 0; i < 8; ++i) acc.f[i] = fmaxf(acc.f[i], 0.0f);

        #pragma unroll
        for (int v = 0; v < 8; ++v) lds[wave][kb + v][col] = acc.f[v];
        asm volatile("s_wait_dscnt 0" ::: "memory");
        AFrag ah;
        #pragma unroll
        for (int h = 0; h < 8; ++h) ah.h[h] = (_Float16)lds[wave][col][kb + h];
        #pragma unroll
        for (int h = 8; h < 16; ++h) ah.h[h] = (_Float16)0.0f;

        #pragma unroll
        for (int j = 0; j < 4; ++j) {
            CFrag c;
            #pragma unroll
            for (int i = 0; i < 8; ++i) c.f[i] = bias2[j];
            c.v = wmma_f16(ah, b2[j], c);
            #pragma unroll
            for (int v = 0; v < 8; ++v)
                xout[(nbase + kb + v) * GNN_D + (unsigned)(j * 16 + col)] =
                    fmaxf(c.f[v], 0.0f);
        }
    }
}

extern "C" void kernel_launch(void* const* d_in, const int* in_sizes, int n_in,
                              void* d_out, int out_size, void* d_ws, size_t ws_size,
                              hipStream_t stream)
{
    (void)in_sizes; (void)n_in; (void)out_size; (void)ws_size;
    const float* x0  = (const float*)d_in[0];
    const int*   ei  = (const int*)  d_in[1];   // [2,E]: row0=src, row1=dst
    const float* Wm1 = (const float*)d_in[2];
    const float* bm1 = (const float*)d_in[3];
    const float* Wm2 = (const float*)d_in[4];
    const float* bm2 = (const float*)d_in[5];
    const float* Wu1 = (const float*)d_in[6];
    const float* bu1 = (const float*)d_in[7];
    const float* Wu2 = (const float*)d_in[8];
    const float* bu2 = (const float*)d_in[9];

    float* bufx = (float*)d_ws;                     // 25.6 MB ping buffer
    float* aggr = bufx + (size_t)GNN_N * GNN_D;     // 25.6 MB aggregation

    const float* xcur = x0;
    for (int l = 0; l < 3; ++l) {
        hipMemsetAsync(aggr, 0, (size_t)GNN_N * GNN_D * sizeof(float), stream);
        edge_msg_scatter<<<4096, 128, 0, stream>>>(
            xcur, ei,
            Wm1 + (size_t)l * GNN_D * GNN_H, bm1 + (size_t)l * GNN_H,
            Wm2 + (size_t)l * GNN_H * GNN_D, bm2 + (size_t)l * GNN_D,
            aggr);
        float* xnext = (l == 2) ? (float*)d_out : bufx;
        node_update<<<1024, 128, 0, stream>>>(
            xcur, aggr,
            Wu1 + (size_t)l * 2 * GNN_D * GNN_H, bu1 + (size_t)l * GNN_H,
            Wu2 + (size_t)l * GNN_H * GNN_D, bu2 + (size_t)l * GNN_D,
            xnext);
        xcur = xnext;
    }
}
